// LSTM2D_46669114638805
// MI455X (gfx1250) — matrix operand order repaired and checked by reading, not compiled, not run on hardware
//
#include <hip/hip_runtime.h>

// ---------- vector types for WMMA ----------
typedef __attribute__((ext_vector_type(16))) __bf16 v16bf;
typedef __attribute__((ext_vector_type(8)))  __bf16 v8bf;
typedef __attribute__((ext_vector_type(8)))  float  v8f;

__device__ __forceinline__ float sigf(float x)      { return 1.0f / (1.0f + __expf(-x)); }
__device__ __forceinline__ float tanh_fast(float x) { float e = __expf(-2.0f * x); return (1.0f - e) / (1.0f + e); }

// 16x32 bf16 A-fragment (contiguous-K order shared with B, not the ISA table order) from LDS row-major, strideE bf16/row.
// lane<16 : M=lane,    K = kk*32 + {0..15}
// lane>=16: M=lane-16, K = kk*32 + {16..31}
__device__ __forceinline__ v16bf load_a_frag(const __bf16* __restrict__ base,
                                             int strideE, int m, int kk, int hi) {
  const __bf16* p = base + m * strideE + kk * 32 + hi * 16;
  v8bf lo = *(const v8bf*)(p);        // ds_read_b128
  v8bf hv = *(const v8bf*)(p + 8);    // ds_read_b128
  v16bf f;
#pragma unroll
  for (int e = 0; e < 8; ++e) { f[e] = lo[e]; f[8 + e] = hv[e]; }
  return f;
}

// 32x16 bf16 B-fragment from LDS copy of an (N x K) row-major weight:
// lane half holds a contiguous 16-K run of row n.
__device__ __forceinline__ v16bf load_b_frag(const __bf16* __restrict__ base,
                                             int strideE, int n, int kk, int hi) {
  const __bf16* p = base + n * strideE + kk * 32 + hi * 16;
  v8bf a = *(const v8bf*)(p);         // ds_read_b128
  v8bf b = *(const v8bf*)(p + 8);     // ds_read_b128
  v16bf f;
#pragma unroll
  for (int e = 0; e < 8; ++e) { f[e] = a[e]; f[8 + e] = b[e]; }
  return f;
}

__device__ __forceinline__ v8f wmma_bf16(v16bf a, v16bf b, v8f c) {
  return __builtin_amdgcn_wmma_f32_16x16x32_bf16(false, a, false, b, (short)0, c, false, false);
}

// ---------------- prep: Wfused = Whh (256x64) @ Wh (64x128) ----------------
// gates = h_inter @ Whh^T = hcat @ (Whh@Wh)^T  (h_inter is linear, only feeds gates)
__global__ void fuse_weights_kernel(const float* __restrict__ Whh,
                                    const float* __restrict__ Wh,
                                    float* __restrict__ Wf) {
  const int idx = blockIdx.x * blockDim.x + threadIdx.x;   // 32768 threads
  const int o = idx >> 7, j = idx & 127;
  float s = 0.0f;
#pragma unroll 8
  for (int k = 0; k < 64; ++k) s += Whh[o * 64 + k] * Wh[k * 128 + j];
  Wf[o * 128 + j] = s;
}

// One workgroup = 4 waves (128 threads) = 16 batch elements; full 256-cell
// serial scan in LDS. Fused gate weights in VGPRs; Wh2 B-frags from LDS.
__global__ __launch_bounds__(128) void lstm2d_scan_kernel(
    const float* __restrict__ X,      // (B,16,16)
    const float* __restrict__ Wx,     // (1,2)
    const float* __restrict__ Wh2,    // (64,128)
    const float* __restrict__ Wih,    // (256,2)
    const float* __restrict__ Wfused, // (256,128)  = Whh @ Wh
    const float* __restrict__ bih,    // (256,)
    const float* __restrict__ bhh,    // (256,)
    const float* __restrict__ Wfc,    // (2,64)
    const float* __restrict__ bfc,    // (2,)
    float* __restrict__ out)          // (B,)
{
  // ---------------- LDS (~94 KB static) ----------------
  __shared__ __align__(16) __bf16 hcat[16 * 128];      // [h_prev | h_up]  A operand
  __shared__ __align__(16) __bf16 ccat[16 * 128];      // [c_prev | c_up]  A operand
  __shared__ __align__(16) float  hf32[16 * 68];       // h_new f32 (padded) for fc head
  __shared__ __align__(16) __bf16 wh2lds[64 * 128];    // Wh2 bf16 (N x K) for B-frags
  __shared__ float xhbuf[16];                          // per-batch-row x embedding
  __shared__ float wfcs[130];                          // Wfc (128) + bfc (2)
  __shared__ float pacc[16 * 2 * 4];                   // fc partial dots
  __shared__ __align__(16) __bf16 row_h[16 * 16 * 64]; // previous image-row h by column
  __shared__ __align__(16) __bf16 row_c[16 * 16 * 64]; // previous image-row c by column

  const int tid  = threadIdx.x;
  const int wave = tid >> 5;
  const int lane = tid & 31;
  const int m16  = lane & 15;
  const int hi   = lane >> 4;

  // ---------------- one-time setup ----------------
  if (tid < 128) wfcs[tid] = Wfc[tid];
  if (tid < 2)   wfcs[128 + tid] = bfc[tid];
  for (int i = tid; i < 64 * 128; i += 128) wh2lds[i] = (__bf16)Wh2[i];

  // Fused gate-weight B fragments in registers (16 frags = 128 VGPRs).
  v16bf bWg[4][4];
  {
    const int n = wave * 16 + m16;
#pragma unroll
    for (int g = 0; g < 4; ++g) {
      const int ng = g * 64 + n;
#pragma unroll
      for (int kk = 0; kk < 4; ++kk) {
        const float* p = Wfused + ng * 128 + kk * 32 + hi * 16;
        v16bf f;
#pragma unroll
        for (int e = 0; e < 16; ++e) f[e] = (__bf16)p[e];
        bWg[g][kk] = f;
      }
    }
  }

  // Folded gate biases: gates[o] = hWf[o] + ba0[o] + xh * ba1[o]
  float ba0[4], ba1[4];
  {
    const int j = wave * 16 + m16;
#pragma unroll
    for (int g = 0; g < 4; ++g) {
      const int o = g * 64 + j;
      ba0[g] = Wih[o * 2 + 1] + bih[o] + bhh[o];
      ba1[g] = Wih[o * 2 + 0] - Wih[o * 2 + 1];
    }
  }
  const float wx0 = Wx[0], wx1 = Wx[1];
  __syncthreads();

  float lp = 0.0f, aa = 0.0f;   // wave0 lanes<16 only

  // ---------------- 256-step serial scan ----------------
#pragma unroll 1
  for (int step = 0; step < 256; ++step) {
    const int r = step >> 4;
    const int t = step & 15;
    const int c = (r & 1) ? (15 - t) : t;    // boustrophedon column

    // (1) wave0 lanes<16: scalar x embedding for this cell
    float x_cur = 0.0f;
    if (tid < 16) {
      const long xb = ((long)(blockIdx.x * 16 + tid)) * 256 + (long)r * 16;
      float xp = 0.0f, xu = 0.0f;
      if (t > 0) { const int cp = (r & 1) ? (16 - t) : (t - 1); xp = X[xb + cp]; }
      if (r > 0) { xu = X[xb - 16 + c]; }
      x_cur = X[xb + c];
      xhbuf[tid] = (xp * wx0 + xu * wx1 + 1.0f) * 0.5f;
    }

    // (2) stage h_up/c_up into hcat/ccat[:,64:128]; zero carries at row start
    {
      const int row = tid >> 3, chunk = tid & 7;
      __bf16* dh = &hcat[row * 128 + 64 + chunk * 8];
      __bf16* dc = &ccat[row * 128 + 64 + chunk * 8];
      if (r == 0) {
        v8bf z = {};
        *(v8bf*)dh = z; *(v8bf*)dc = z;
      } else {
        *(v8bf*)dh = *(const v8bf*)&row_h[(c * 16 + row) * 64 + chunk * 8];
        *(v8bf*)dc = *(const v8bf*)&row_c[(c * 16 + row) * 64 + chunk * 8];
      }
      if (t == 0) {           // h_prev = c_prev = 0 at start of each row
        v8bf z = {};
        *(v8bf*)&hcat[row * 128 + chunk * 8] = z;
        *(v8bf*)&ccat[row * 128 + chunk * 8] = z;
      }
    }
    __syncthreads();                                   // barrier A

    // (3) fused GEMMs over K=128: 16 gate WMMAs (4 independent chains) + c_inter
    v8f g4[4] = {v8f{}, v8f{}, v8f{}, v8f{}};
    v8f cint  = {};
#pragma unroll
    for (int kk = 0; kk < 4; ++kk) {
      v16bf ah = load_a_frag(hcat, 128, m16, kk, hi);
#pragma unroll
      for (int g = 0; g < 4; ++g) g4[g] = wmma_bf16(ah, bWg[g][kk], g4[g]);
      v16bf ac = load_a_frag(ccat, 128, m16, kk, hi);
      v16bf bc = load_b_frag(wh2lds, 128, wave * 16 + m16, kk, hi);
      cint = wmma_bf16(ac, bc, cint);
    }
    __syncthreads();                                   // barrier B1 (reads done before writes)

    // (4) elementwise LSTM update (wave-local: c_inter tile == gate columns)
    float xh8[8];
#pragma unroll
    for (int v = 0; v < 8; ++v) xh8[v] = xhbuf[hi * 8 + v];
    float hnew[8], cnew[8];
#pragma unroll
    for (int v = 0; v < 8; ++v) {
      const float xh = xh8[v];
      const float ig = g4[0][v] + ba0[0] + xh * ba1[0];
      const float fg = g4[1][v] + ba0[1] + xh * ba1[1];
      const float gg = g4[2][v] + ba0[2] + xh * ba1[2];
      const float og = g4[3][v] + ba0[3] + xh * ba1[3];
      const float cn = sigf(fg) * cint[v] + sigf(ig) * tanh_fast(gg);
      const float hn = sigf(og) * tanh_fast(cn);
      hnew[v] = hn; cnew[v] = cn;
    }

    // (5) restage h_new/c_new for next step + next image row + fc head
    {
      const int j = wave * 16 + m16;
#pragma unroll
      for (int v = 0; v < 8; ++v) {
        const int row = v + hi * 8;
        hcat[row * 128 + j] = (__bf16)hnew[v];
        ccat[row * 128 + j] = (__bf16)cnew[v];
        hf32[row * 68 + j]  = hnew[v];
        row_h[(c * 16 + row) * 64 + j] = (__bf16)hnew[v];
        row_c[(c * 16 + row) * 64 + j] = (__bf16)cnew[v];
      }
    }
    __syncthreads();                                   // barrier B2

    // (6) fc head partial dots: thread = (batch row m, output o, j-quarter q)
    {
      const int m = tid & 15, o = (tid >> 4) & 1, q = tid >> 5;
      float p = 0.0f;
#pragma unroll
      for (int e = 0; e < 16; ++e) {
        const int jj = q * 16 + e;
        p += hf32[m * 68 + jj] * wfcs[o * 64 + jj];
      }
      pacc[(m * 2 + o) * 4 + q] = p;
    }
    __syncthreads();                                   // barrier C

    // (7) wave0 lanes<16: finalize log_softmax + log-prob accumulation
    if (tid < 16) {
      float y0 = wfcs[128], y1 = wfcs[129];
#pragma unroll
      for (int q = 0; q < 4; ++q) {
        y0 += pacc[(tid * 2 + 0) * 4 + q];
        y1 += pacc[(tid * 2 + 1) * 4 + q];
      }
      const float mx  = fmaxf(y0, y1);
      const float lse = mx + __logf(__expf(y0 - mx) + __expf(y1 - mx));
      const float mask = (1.0f + x_cur) * 0.5f;
      float contrib = (y0 - lse) * mask + (y1 - lse) * (1.0f - mask);
      aa += mask;
      if (step == 255) {                 // cell (15,0): recovery correction
        if (aa == 1.0f) contrib *= (1.0f - mask);
      }
      lp += contrib;
    }
  }

  if (tid < 16) out[blockIdx.x * 16 + tid] = lp;
}

extern "C" void kernel_launch(void* const* d_in, const int* in_sizes, int n_in,
                              void* d_out, int out_size, void* d_ws, size_t ws_size,
                              hipStream_t stream) {
  (void)n_in; (void)ws_size; (void)out_size;
  const float* X   = (const float*)d_in[0];
  const float* Wx  = (const float*)d_in[1];
  const float* Wh  = (const float*)d_in[2];
  const float* Wh2 = (const float*)d_in[3];
  const float* Wih = (const float*)d_in[4];
  const float* Whh = (const float*)d_in[5];
  const float* bih = (const float*)d_in[6];
  const float* bhh = (const float*)d_in[7];
  const float* Wfc = (const float*)d_in[8];
  const float* bfc = (const float*)d_in[9];
  float* out    = (float*)d_out;
  float* Wfused = (float*)d_ws;        // 256*128*4 = 128 KB scratch

  // Fuse Whh@Wh once per call (same stream => ordered before the scan kernel).
  fuse_weights_kernel<<<128, 256, 0, stream>>>(Whh, Wh, Wfused);

  const int B = in_sizes[0] / 256;     // (B,16,16)
  const int nblocks = B / 16;          // 16 batch elements per workgroup
  lstm2d_scan_kernel<<<nblocks, 128, 0, stream>>>(
      X, Wx, Wh2, Wih, Wfused, bih, bhh, Wfc, bfc, out);
}
